// LearnedImportanceSampler_18940805775799
// MI455X (gfx1250) — matrix operand order repaired and checked by reading, not compiled, not run on hardware
//
#include <hip/hip_runtime.h>
#include <hip/hip_bf16.h>

typedef __attribute__((ext_vector_type(16))) _Float16 v16h;
typedef __attribute__((ext_vector_type(8)))  _Float16 v8h;
typedef __attribute__((ext_vector_type(8)))  float    v8f;

#define Bn 2
#define SQ 256
#define SK 512
#define Ed 512
#define Hd 256

// ---------------------------------------------------------------------------
// Transpose + convert: src f32 [K][N] -> dst f16 [N][K]
// ---------------------------------------------------------------------------
__global__ void tcvt_f32_to_f16T(const float* __restrict__ src,
                                 _Float16* __restrict__ dst, int K, int N) {
    int idx = blockIdx.x * 256 + threadIdx.x;
    if (idx < K * N) {
        int n = idx / K, k = idx % K;
        dst[idx] = (_Float16)src[k * N + n];
    }
}

// ---------------------------------------------------------------------------
// Generic WMMA GEMM: out[M][N] = act(A[M][K] @ W + bias), W given as f16 WT[N][K].
// One wave (32 threads) per block computes a 16x64 tile. M%16==0, N%64==0, K%32==0.
// ---------------------------------------------------------------------------
template<bool HASBIAS, bool RELU>
__launch_bounds__(32)
__global__ void gemm_rowtile(const float* __restrict__ A, const _Float16* __restrict__ WT,
                             const float* __restrict__ bias, float* __restrict__ out,
                             int M, int K, int N) {
    int ntn  = N / 64;
    int mb   = blockIdx.x / ntn;
    int nb   = blockIdx.x % ntn;
    int lane = threadIdx.x;
    int r16  = lane & 15, hi = lane >> 4;
    int m0   = mb * 16;

    v8f acc[4] = {};
    const float* arow = A + (size_t)(m0 + r16) * K;

    for (int kc = 0; kc < K / 32; ++kc) {
        int ka0 = kc * 32 + hi * 16;
        v16h a;
#pragma unroll
        for (int i = 0; i < 8; ++i) a[i]     = (_Float16)arow[ka0 + i];
#pragma unroll
        for (int i = 0; i < 8; ++i) a[i + 8] = (_Float16)arow[ka0 + 8 + i];
        int kb = kc * 32 + hi * 16;
#pragma unroll
        for (int nt = 0; nt < 4; ++nt) {
            int n = nb * 64 + nt * 16 + r16;
            const v8h* bp = (const v8h*)(WT + (size_t)n * K + kb);
            v8h b0v = bp[0], b1v = bp[1];
            v16h bm;
#pragma unroll
            for (int i = 0; i < 8; ++i) { bm[i] = b0v[i]; bm[i + 8] = b1v[i]; }
            acc[nt] = __builtin_amdgcn_wmma_f32_16x16x32_f16(
                false, a, false, bm, (short)0, acc[nt], false, false);
        }
    }
#pragma unroll
    for (int nt = 0; nt < 4; ++nt) {
        int n = nb * 64 + nt * 16 + r16;
        float bv = HASBIAS ? bias[n] : 0.f;
#pragma unroll
        for (int j = 0; j < 8; ++j) {
            int m = m0 + j + 8 * hi;            // C layout: vgpr j, lane half -> row
            float v = acc[nt][j] + bv;
            if (RELU) v = fmaxf(v, 0.f);
            out[(size_t)m * N + n] = v;
        }
    }
}

// ---------------------------------------------------------------------------
// Async staging of a 64-row x 256-half W panel (32 KB = 4 n-tiles) into LDS.
// Per thread: 128 bytes via eight GLOBAL_LOAD_ASYNC_TO_LDS_B128 (ASYNCcnt).
// ---------------------------------------------------------------------------
__device__ __forceinline__ void stage_async_step(const _Float16* __restrict__ WT, int step,
                                                 _Float16* Btb, int tid) {
    int n  = tid >> 2;                 // 0..63: row within the 64-row panel
    int ks = (tid & 3) * 64;           // 64 halfs = 128 B per thread
    const _Float16* g = WT + (size_t)(step * 64 + n) * 256 + ks;
    unsigned lds = (unsigned)(uintptr_t)(Btb + n * 256 + ks);   // low 32b = LDS offset
    unsigned long long ga = (unsigned long long)(uintptr_t)g;
    asm volatile(
        "global_load_async_to_lds_b128 %0, %1, off\n\t"
        "global_load_async_to_lds_b128 %0, %1, off offset:16\n\t"
        "global_load_async_to_lds_b128 %0, %1, off offset:32\n\t"
        "global_load_async_to_lds_b128 %0, %1, off offset:48\n\t"
        "global_load_async_to_lds_b128 %0, %1, off offset:64\n\t"
        "global_load_async_to_lds_b128 %0, %1, off offset:80\n\t"
        "global_load_async_to_lds_b128 %0, %1, off offset:96\n\t"
        "global_load_async_to_lds_b128 %0, %1, off offset:112"
        :: "v"(lds), "v"(ga) : "memory");
}

// ---------------------------------------------------------------------------
// Load one 16x16x32 B fragment (lane half picks K sub-range) from an LDS row.
// ---------------------------------------------------------------------------
__device__ __forceinline__ v16h load_bfrag(const _Float16* brow, int kc, int hi) {
    int kb = kc * 32 + hi * 16;
    v8h b0 = *(const v8h*)(brow + kb);
    v8h b1 = *(const v8h*)(brow + kb + 8);
    v16h r;
#pragma unroll
    for (int i = 0; i < 8; ++i) { r[i] = b0[i]; r[i + 8] = b1[i]; }
    return r;
}

// ---------------------------------------------------------------------------
// Per-wave 16x256 x 256x256 GEMM layer. A fragments hoisted to registers,
// W panels (4 n-tiles each) double-buffered in LDS via pipelined async copies.
// Compute: flat 32-fragment loop, B prefetch distance 2 (3 live fragments),
// accumulating WMMAs chain back-to-back (D->C carries no hazard penalty).
// ---------------------------------------------------------------------------
__device__ __forceinline__ void wave_gemm_256(const _Float16* __restrict__ WT,
                                              v8f acc[16],
                                              const _Float16* Aw,     // per-wave [16][256]
                                              _Float16* Bt0, _Float16* Bt1,
                                              int tid, int lane) {
    int r16 = lane & 15, hi = lane >> 4;

    // preload all 8 A fragments (16 ds_load_b128 total, reused for all 16 n-tiles)
    const _Float16* arow = Aw + r16 * 256;
    v16h afr[8];
#pragma unroll
    for (int kc = 0; kc < 8; ++kc) {
        v8h a0 = *(const v8h*)(arow + kc * 32 + hi * 16);
        v8h a1 = *(const v8h*)(arow + kc * 32 + hi * 16 + 8);
#pragma unroll
        for (int i = 0; i < 8; ++i) { afr[kc][i] = a0[i]; afr[kc][i + 8] = a1[i]; }
    }

    stage_async_step(WT, 0, Bt0, tid);     // prologue: panel 0 in flight
#pragma unroll
    for (int step = 0; step < 4; ++step) {
        _Float16* cur = (step & 1) ? Bt1 : Bt0;
        _Float16* nxt = (step & 1) ? Bt0 : Bt1;
        __syncthreads();                   // all waves done reading `nxt` buffer
        if (step < 3) {
            stage_async_step(WT, step + 1, nxt, tid);
            asm volatile("s_wait_asynccnt 0x8" ::: "memory");   // panel `step` landed
        } else {
            asm volatile("s_wait_asynccnt 0x0" ::: "memory");
        }
        __syncthreads();                   // panel visible to all waves

        // 32 fragments = 4 n-tiles x 8 k-chunks, prefetch distance 2
        const _Float16* crow = cur + r16 * 256;     // single base reg; rest folds to imm
        v16h bf0 = load_bfrag(crow, 0, hi);
        v16h bf1 = load_bfrag(crow, 1, hi);
        v8f c = {};
#pragma unroll
        for (int u = 0; u < 32; ++u) {
            int t = u >> 3, kc = u & 7;
            v16h bnext;
            if (u < 30) {
                int u2 = u + 2;
                bnext = load_bfrag(crow + (u2 >> 3) * (16 * 256), u2 & 7, hi);
            }
            c = __builtin_amdgcn_wmma_f32_16x16x32_f16(
                false, afr[kc], false, bf0, (short)0, c, false, false);
            if (kc == 7) {
                acc[step * 4 + t] = c;
                c = (v8f){0.f, 0.f, 0.f, 0.f, 0.f, 0.f, 0.f, 0.f};
            }
            bf0 = bf1;
            if (u < 30) bf1 = bnext;
        }
    }
}

// ---------------------------------------------------------------------------
// Fused kernel: per workgroup = 128 (q,k) rows (one q, 128 consecutive k).
// h0 = relu(qp+kp+b0) -> LDS; h1 = relu(h0@W1+b1); h2 = relu(h1@W2+b2);
// logits = h2@Wf + bf.  Variance path fused in phase 0.
// ---------------------------------------------------------------------------
__launch_bounds__(256)
__global__ void fused_importance(const float* __restrict__ qp, const float* __restrict__ kp,
                                 const float* __restrict__ qv, const float* __restrict__ kv,
                                 const _Float16* __restrict__ W1T, const _Float16* __restrict__ W2T,
                                 const float* __restrict__ b0, const float* __restrict__ b1,
                                 const float* __restrict__ b2, const float* __restrict__ Wf,
                                 const float* __restrict__ bfp, const float* __restrict__ bv1,
                                 const float* __restrict__ Wv2, const float* __restrict__ bv2,
                                 float* __restrict__ out_logits, float* __restrict__ out_var) {
    // ONE shared array => layout under our control.  B double-buffer occupies
    // LDS [0, 64 KB) so every hot-loop fragment address is base-reg + 16-bit
    // immediate; per-wave A panels live above 64 KB (base computed once).
    __shared__ __align__(16) _Float16 smem[65536];     // 128 KB total
    _Float16* Bt0 = smem;                              // [64][256]  32 KB @ 0x0000
    _Float16* Bt1 = smem + 64 * 256;                   // [64][256]  32 KB @ 0x8000
    _Float16* Apanels = smem + 2 * 64 * 256;           // 8 x [16][256] 64 KB @ 0x10000

    int tid = threadIdx.x;
    int w = tid >> 5, lane = tid & 31;
    int r16 = lane & 15, hi = lane >> 4;
    _Float16* Aw = Apanels + w * (16 * 256);           // this wave's A panel

    long row0 = (long)blockIdx.x * 128;
    int b   = (int)(row0 / (SQ * SK));
    int rem = (int)(row0 % (SQ * SK));
    int q   = rem / SK;
    int k0  = rem % SK;

    // ---------------- Phase 0: h0 panel + variance ----------------
    const float* qpr = qp + ((size_t)b * SQ + q) * Hd;
    const float* qvr = qv + ((size_t)b * SQ + q) * Hd;
    int c0 = lane * 8;
    float qpc[8], qvc[8], b0c[8], bv1c[8], wv2c[8];
#pragma unroll
    for (int i = 0; i < 8; ++i) {
        qpc[i]  = qpr[c0 + i];  qvc[i]  = qvr[c0 + i];
        b0c[i]  = b0[c0 + i];   bv1c[i] = bv1[c0 + i];
        wv2c[i] = Wv2[c0 + i];
    }
    float bvv2 = bv2[0];

    for (int rr = 0; rr < 16; ++rr) {
        int krow = k0 + w * 16 + rr;
        const float* kpr = kp + ((size_t)b * SK + krow) * Hd;
        const float* kvr = kv + ((size_t)b * SK + krow) * Hd;
        float vacc = 0.f;
        v8h h8;
#pragma unroll
        for (int i = 0; i < 8; ++i) {
            float h0 = fmaxf(qpc[i] + kpr[c0 + i] + b0c[i], 0.f);
            h8[i] = (_Float16)h0;
            float vv = fmaxf(qvc[i] + kvr[c0 + i] + bv1c[i], 0.f);
            vacc += vv * wv2c[i];
        }
        *(v8h*)(Aw + rr * 256 + c0) = h8;
        for (int off = 16; off >= 1; off >>= 1) vacc += __shfl_xor(vacc, off, 32);
        if (lane == 0) {
            float x = vacc + bvv2;
            float sp = (x > 20.f) ? x : log1pf(expf(x));   // softplus
            out_var[((size_t)b * SQ + q) * SK + krow] = sp;
        }
    }

    v8f acc[16];

    // ---------------- Layer 1 ----------------
    wave_gemm_256(W1T, acc, Aw, Bt0, Bt1, tid, lane);
    // relu + bias, write h1 back to A panel as f16
    {
        float bias1[16];
#pragma unroll
        for (int nt = 0; nt < 16; ++nt) bias1[nt] = b1[nt * 16 + r16];
        _Float16* wb = Aw + (8 * hi) * 256 + r16;
#pragma unroll
        for (int nt = 0; nt < 16; ++nt)
#pragma unroll
            for (int j = 0; j < 8; ++j) {
                float v = fmaxf(acc[nt][j] + bias1[nt], 0.f);
                wb[j * 256 + nt * 16] = (_Float16)v;
            }
    }

    // ---------------- Layer 2 ----------------
    wave_gemm_256(W2T, acc, Aw, Bt0, Bt1, tid, lane);

    // ---------------- Final dot with Wf ----------------
    float p[8] = {0.f, 0.f, 0.f, 0.f, 0.f, 0.f, 0.f, 0.f};
    {
        float bias2[16], wfl[16];
#pragma unroll
        for (int nt = 0; nt < 16; ++nt) {
            int n = nt * 16 + r16;
            bias2[nt] = b2[n]; wfl[nt] = Wf[n];
        }
#pragma unroll
        for (int nt = 0; nt < 16; ++nt) {
            float wf = wfl[nt];
#pragma unroll
            for (int j = 0; j < 8; ++j)
                p[j] += fmaxf(acc[nt][j] + bias2[nt], 0.f) * wf;
        }
    }
    for (int off = 8; off >= 1; off >>= 1)
#pragma unroll
        for (int j = 0; j < 8; ++j) p[j] += __shfl_xor(p[j], off, 32);
    if (r16 == 0) {
        size_t base = ((size_t)b * SQ + q) * SK + k0 + w * 16 + 8 * hi;
        float bff = bfp[0];
#pragma unroll
        for (int j = 0; j < 8; ++j) out_logits[base + j] = p[j] + bff;
    }
}

// ---------------------------------------------------------------------------
extern "C" void kernel_launch(void* const* d_in, const int* in_sizes, int n_in,
                              void* d_out, int out_size, void* d_ws, size_t ws_size,
                              hipStream_t stream) {
    const float* query = (const float*)d_in[0];
    const float* key   = (const float*)d_in[1];
    const float* Wqe = (const float*)d_in[2];  const float* bqe = (const float*)d_in[3];
    const float* Wke = (const float*)d_in[4];  const float* bke = (const float*)d_in[5];
    const float* W0  = (const float*)d_in[6];  const float* b0  = (const float*)d_in[7];
    const float* W1  = (const float*)d_in[8];  const float* b1  = (const float*)d_in[9];
    const float* W2  = (const float*)d_in[10]; const float* b2  = (const float*)d_in[11];
    const float* Wf  = (const float*)d_in[12]; const float* bf  = (const float*)d_in[13];
    const float* Wv1 = (const float*)d_in[14]; const float* bv1 = (const float*)d_in[15];
    const float* Wv2 = (const float*)d_in[16]; const float* bv2 = (const float*)d_in[17];

    // workspace layout (~6.0 MB total)
    _Float16* WqeT  = (_Float16*)d_ws;        // [256][512]
    _Float16* WkeT  = WqeT  + 256 * 512;      // [256][512]
    _Float16* W0qT  = WkeT  + 256 * 512;      // [256][256]
    _Float16* W0kT  = W0qT  + 256 * 256;
    _Float16* Wv1qT = W0kT  + 256 * 256;
    _Float16* Wv1kT = Wv1qT + 256 * 256;
    _Float16* W1T   = Wv1kT + 256 * 256;
    _Float16* W2T   = W1T   + 256 * 256;
    float* qf  = (float*)(W2T + 256 * 256);   // [512][256]
    float* kf  = qf  + 512  * 256;            // [1024][256]
    float* qpw = kf  + 1024 * 256;            // [512][256]
    float* kpw = qpw + 512  * 256;            // [1024][256]
    float* qvw = kpw + 1024 * 256;            // [512][256]
    float* kvw = qvw + 512  * 256;            // [1024][256]

    // weight transpose/convert (tiny)
    tcvt_f32_to_f16T<<<(512*256 + 255)/256, 256, 0, stream>>>(Wqe, WqeT, 512, 256);
    tcvt_f32_to_f16T<<<(512*256 + 255)/256, 256, 0, stream>>>(Wke, WkeT, 512, 256);
    tcvt_f32_to_f16T<<<(256*256 + 255)/256, 256, 0, stream>>>(W0,            W0qT,  256, 256);
    tcvt_f32_to_f16T<<<(256*256 + 255)/256, 256, 0, stream>>>(W0 + 256*256,  W0kT,  256, 256);
    tcvt_f32_to_f16T<<<(256*256 + 255)/256, 256, 0, stream>>>(Wv1,           Wv1qT, 256, 256);
    tcvt_f32_to_f16T<<<(256*256 + 255)/256, 256, 0, stream>>>(Wv1 + 256*256, Wv1kT, 256, 256);
    tcvt_f32_to_f16T<<<(256*256 + 255)/256, 256, 0, stream>>>(W1, W1T, 256, 256);
    tcvt_f32_to_f16T<<<(256*256 + 255)/256, 256, 0, stream>>>(W2, W2T, 256, 256);

    // encoders: qf = relu(query@Wqe+bqe), kf = relu(key@Wke+bke)
    gemm_rowtile<true,  true ><<<(512 /16)*(256/64), 32, 0, stream>>>(query, WqeT, bqe, qf, 512,  512, 256);
    gemm_rowtile<true,  true ><<<(1024/16)*(256/64), 32, 0, stream>>>(key,   WkeT, bke, kf, 1024, 512, 256);
    // partial projections (no bias/relu)
    gemm_rowtile<false, false><<<(512 /16)*(256/64), 32, 0, stream>>>(qf, W0qT,  nullptr, qpw, 512,  256, 256);
    gemm_rowtile<false, false><<<(1024/16)*(256/64), 32, 0, stream>>>(kf, W0kT,  nullptr, kpw, 1024, 256, 256);
    gemm_rowtile<false, false><<<(512 /16)*(256/64), 32, 0, stream>>>(qf, Wv1qT, nullptr, qvw, 512,  256, 256);
    gemm_rowtile<false, false><<<(1024/16)*(256/64), 32, 0, stream>>>(kf, Wv1kT, nullptr, kvw, 1024, 256, 256);

    // fused per-pair MLP (the 69 GFLOP part)
    float* out_logits = (float*)d_out;
    float* out_var    = out_logits + (size_t)Bn * SQ * SK;
    fused_importance<<<(Bn * SQ * SK) / 128, 256, 0, stream>>>(
        qpw, kpw, qvw, kvw, W1T, W2T, b0, b1, b2, Wf, bf, bv1, Wv2, bv2,
        out_logits, out_var);
}
